// MeganDecoder_19628000542989
// MI455X (gfx1250) — matrix operand order repaired and checked by reading, not compiled, not run on hardware
//
#include <hip/hip_runtime.h>
#include <stdint.h>
#include <stddef.h>

// ---------------- types / helpers ----------------
typedef unsigned short u16b;                                   // raw bf16 storage
typedef __attribute__((ext_vector_type(16))) __bf16 v16bf;     // WMMA A/B fragment
typedef __attribute__((ext_vector_type(8)))  float  v8f;       // WMMA C/D fragment

#define DEVFN static __device__ __forceinline__

DEVFN float bf2f(u16b h){ union{unsigned u; float f;} x; x.u = ((unsigned)h)<<16; return x.f; }
DEVFN u16b f2bf(float f){ union{float f; unsigned u;} x; x.f=f; unsigned u=x.u;
  unsigned r = u + 0x7FFFu + ((u>>16)&1u); return (u16b)(r>>16); }
// packed bf16x2 add (round-to-nearest-even)
DEVFN unsigned addbf2(unsigned a, unsigned b){
  union{unsigned u; float f;} al, ah, bl, bh;
  al.u = a << 16; ah.u = a & 0xFFFF0000u;
  bl.u = b << 16; bh.u = b & 0xFFFF0000u;
  return ((unsigned)f2bf(al.f + bl.f)) | (((unsigned)f2bf(ah.f + bh.f)) << 16);
}
// orderable-uint encoding of float for atomicMax
DEVFN unsigned fkey(float f){ union{float f; unsigned u;} x; x.f=f;
  return (x.u & 0x80000000u) ? ~x.u : (x.u | 0x80000000u); }
DEVFN float fdec(unsigned k){ union{unsigned u; float f;} x;
  x.u = (k & 0x80000000u) ? (k & 0x7FFFFFFFu) : ~k; return x.f; }
DEVFN int imin(int a, int b){ return a < b ? a : b; }

// ---------------- problem constants ----------------
static const int cN = 10000, cE = 30000, cES = 20000, cD = 512, cL = 4;
static const int FCH = 256, NAo = 100, NBo = 50, cBA = 32;

// ---------------- WMMA GEMM ----------------
// Block: 256 threads (8 waves), tile 128(M) x 64(N), K-step 32, bf16 A/B, f32 acc.
// Out-of-range A rows / B cols are CLAMPED (not zeroed): the garbage lands only in
// ghost accumulator rows/cols that the masked epilogue never stores (K is exact).
// A-source modes:
//  0 DIRECT : A0[row*K + k]
//  1 G3     : cols [0,segW): A0[idx0[r]], [segW,2segW): A0[idx1[r]], [2segW,3segW): A1[idx2?idx2[r]:r]
//  2 G2S    : cols [0,segW): A0[idx0[r]] + A0[idx1[r]], [segW,2segW): A1[idx2?idx2[r]:r]
// segW must be a multiple of 32. K must be a multiple of 32.
template<int MODE>
__global__ __launch_bounds__(256)
void gemm_wmma(const u16b* __restrict__ A0, const u16b* __restrict__ A1,
               const int* __restrict__ idx0, const int* __restrict__ idx1,
               const int* __restrict__ idx2,
               const u16b* __restrict__ B, const float* __restrict__ bias,
               float* __restrict__ outF, u16b* __restrict__ outB,
               int M, int N, int K, int segW, int act)
{
  __shared__ __align__(16) u16b ldsA[128 * 40];   // 32 data + 8 pad per row
  __shared__ __align__(16) u16b ldsB[64 * 40];    // transposed: [n][k], 32 data + 8 pad

  const int tid  = threadIdx.x;
  const int lane = tid & 31;
  const int wave = tid >> 5;
  const int tM = blockIdx.y * 128;
  const int tN = blockIdx.x * 64;
  const int wm = (wave >> 1) * 32;   // 4 M-groups of 32
  const int wn = (wave & 1) * 32;    // 2 N-groups of 32

  v8f zero8 = {0.f,0.f,0.f,0.f,0.f,0.f,0.f,0.f};
  v8f acc[2][2];
  acc[0][0]=zero8; acc[0][1]=zero8; acc[1][0]=zero8; acc[1][1]=zero8;

  const int rowL = tid >> 1;          // 0..127 (A tile row per 2 threads)
  const int hk   = tid & 1;           // 16-col half
  const int grc  = imin(tM + rowL, M - 1);   // clamped row (ghost-safe)
  const int bk   = tid >> 3;          // 0..31  (B tile row)
  const int bn   = (tid & 7) * 8;     // B col group of 8

  const int hl  = lane >> 4;          // half-wave
  const int l16 = lane & 15;

  // block-uniform: whole 64-col B tile in bounds & 16B-aligned vector loads legal
  const bool nFull = ((N & 7) == 0) && (tN + 64 <= N);
  const bool mFull = (tM + 128 <= M);

  // hoisted gather row bases (loop-invariant for MODE != 0)
  const u16b* rowP0 = nullptr;  // seg0 base
  const u16b* rowP1 = nullptr;  // seg1 base (G3) / second seg0 operand (G2S)
  const u16b* rowP2 = nullptr;  // tail segment base
  if (MODE != 0) {
    rowP0 = A0 + (size_t)idx0[grc] * segW;
    if (MODE == 1) rowP1 = A0 + (size_t)idx1[grc] * segW;
    else           rowP1 = A0 + (size_t)idx1[grc] * segW;   // G2S second operand
    const int rr = idx2 ? idx2[grc] : grc;
    rowP2 = A1 + (size_t)rr * segW;
  }

  for (int k0 = 0; k0 < K; k0 += 32) {
    // ---- A tile -> LDS (row-major, padded) ----
    {
      const int kc = k0 + hk * 16;    // 16 contiguous cols, stays inside one segment
      uint4 u0, u1;
      if (MODE == 0) {
        const u16b* p = A0 + (size_t)grc * K + kc;
        u0 = *(const uint4*)p; u1 = *(const uint4*)(p + 8);
        if (k0 + 32 < K) __builtin_prefetch(p + 32, 0, 1);
      } else {
        const int seg = kc / segW;
        const int ko  = kc - seg * segW;
        if (MODE == 1) {
          const u16b* p = (seg == 0) ? (rowP0 + ko) : (seg == 1) ? (rowP1 + ko) : (rowP2 + ko);
          u0 = *(const uint4*)p; u1 = *(const uint4*)(p + 8);
          __builtin_prefetch(p + 32, 0, 1);
        } else { // MODE 2: gather2 + sum (vectorized)
          if (seg == 0) {
            const u16b* p = rowP0 + ko;
            const u16b* q = rowP1 + ko;
            const uint4 p0 = *(const uint4*)p, p1 = *(const uint4*)(p + 8);
            const uint4 q0 = *(const uint4*)q, q1 = *(const uint4*)(q + 8);
            u0.x = addbf2(p0.x, q0.x); u0.y = addbf2(p0.y, q0.y);
            u0.z = addbf2(p0.z, q0.z); u0.w = addbf2(p0.w, q0.w);
            u1.x = addbf2(p1.x, q1.x); u1.y = addbf2(p1.y, q1.y);
            u1.z = addbf2(p1.z, q1.z); u1.w = addbf2(p1.w, q1.w);
          } else {
            const u16b* p = rowP2 + ko;
            u0 = *(const uint4*)p; u1 = *(const uint4*)(p + 8);
          }
        }
      }
      u16b* d = &ldsA[rowL * 40 + hk * 16];
      *(uint4*)d = u0; *(uint4*)(d + 8) = u1;
    }
    // ---- B tile -> LDS transposed (Bt[n][k]) ----
    {
      const int kk = k0 + bk;
      if (nFull) {
        const u16b* bp = B + (size_t)kk * N + tN + bn;
        const uint4 u = *(const uint4*)bp;
        const u16b* bv = (const u16b*)&u;
        #pragma unroll
        for (int j = 0; j < 8; ++j) ldsB[(bn + j) * 40 + bk] = bv[j];
        if (k0 + 32 < K) __builtin_prefetch(bp + (size_t)32 * N, 0, 1);
      } else {
        #pragma unroll
        for (int j = 0; j < 8; ++j) {
          const int n = imin(tN + bn + j, N - 1);   // clamped (ghost-safe)
          ldsB[(bn + j) * 40 + bk] = B[(size_t)kk * N + n];
        }
      }
    }
    __syncthreads();

    // ---- fragments per ISA 16-bit layouts + WMMA ----
    union Frag { uint4 u[2]; v16bf v; };
    Frag fa[2], fb[2];
    #pragma unroll
    for (int i = 0; i < 2; ++i) {
      // A 16x32: lane<16 holds K0..7 & K16..23; lane>=16 holds K8..15 & K24..31
      const u16b* p = &ldsA[(wm + i * 16 + l16) * 40 + hl * 8];
      fa[i].u[0] = *(const uint4*)p;
      fa[i].u[1] = *(const uint4*)(p + 16);
    }
    #pragma unroll
    for (int j = 0; j < 2; ++j) {
      // B 32x16 (from Bt): lane<16 holds K0..7 & K16..23 of col n; lane>=16 holds K8..15 & K24..31
      const u16b* p = &ldsB[(wn + j * 16 + l16) * 40 + hl * 8];
      fb[j].u[0] = *(const uint4*)p;
      fb[j].u[1] = *(const uint4*)(p + 16);
    }
    #pragma unroll
    for (int i = 0; i < 2; ++i)
      #pragma unroll
      for (int j = 0; j < 2; ++j)
        acc[i][j] = __builtin_amdgcn_wmma_f32_16x16x32_bf16(
            false, fa[i].v, false, fb[j].v, (short)0, acc[i][j], false, false);
    __syncthreads();
  }

  // ---- epilogue: bias + act + dual store ----
  if (mFull && nFull) {
    #pragma unroll
    for (int i = 0; i < 2; ++i) {
      #pragma unroll
      for (int j = 0; j < 2; ++j) {
        const int col = tN + wn + j * 16 + l16;
        const float bv = bias ? bias[col] : 0.f;
        #pragma unroll
        for (int r = 0; r < 8; ++r) {
          const int row = tM + wm + i * 16 + (hl ? 8 + r : r);
          float v = acc[i][j][r] + bv;
          if (act == 1) v = v > 0.f ? v : 0.f;
          else if (act == 2) v = tanhf(v);
          if (outF) outF[(size_t)row * N + col] = v;
          if (outB) outB[(size_t)row * N + col] = f2bf(v);
        }
      }
    }
  } else {
    #pragma unroll
    for (int i = 0; i < 2; ++i) {
      #pragma unroll
      for (int j = 0; j < 2; ++j) {
        const int col = tN + wn + j * 16 + l16;
        if (col >= N) continue;
        const float bv = bias ? bias[col] : 0.f;
        #pragma unroll
        for (int r = 0; r < 8; ++r) {
          const int row = tM + wm + i * 16 + (hl ? 8 + r : r);
          if (row >= M) continue;
          float v = acc[i][j][r] + bv;
          if (act == 1) v = v > 0.f ? v : 0.f;
          else if (act == 2) v = tanhf(v);
          if (outF) outF[(size_t)row * N + col] = v;
          if (outB) outB[(size_t)row * N + col] = f2bf(v);
        }
      }
    }
  }
}

// ---------------- elementwise / graph kernels ----------------
__global__ __launch_bounds__(256) void k_fill_f32(float* p, float v, size_t n){
  size_t i = (size_t)blockIdx.x * 256 + threadIdx.x; if (i < n) p[i] = v; }
__global__ __launch_bounds__(256) void k_fill_u32(unsigned* p, unsigned v, size_t n){
  size_t i = (size_t)blockIdx.x * 256 + threadIdx.x; if (i < n) p[i] = v; }
__global__ __launch_bounds__(256) void k_f2bf(const float* s, u16b* d, size_t n){
  size_t i = (size_t)blockIdx.x * 256 + threadIdx.x; if (i < n) d[i] = f2bf(s[i]); }
__global__ __launch_bounds__(256) void k_relu(float* p, size_t n){
  size_t i = (size_t)blockIdx.x * 256 + threadIdx.x; if (i < n){ float v=p[i]; p[i]=v>0.f?v:0.f; } }
__global__ __launch_bounds__(256) void k_build_edges(const int* eidx, int* es, int* ed, int n){
  int i = blockIdx.x * 256 + threadIdx.x; if (i < n){ es[i]=eidx[2*i]; ed[i]=eidx[2*i+1]; } }
__global__ __launch_bounds__(256) void k_build_sub(const int* eidx, const int* sub, int* ss, int* sd, int n){
  int i = blockIdx.x * 256 + threadIdx.x; if (i < n){ int e=sub[i]; ss[i]=eidx[2*e]; sd[i]=eidx[2*e+1]; } }

// scores = h2(bf16, Mx512) @ w3(512x8,f32) + b3 ; atomicMax of segment key by src
__global__ __launch_bounds__(256) void k_scores(const u16b* __restrict__ h2,
    const float* __restrict__ w3, const float* __restrict__ b3,
    const int* __restrict__ srcI, float* __restrict__ scores,
    unsigned* __restrict__ segK, int M)
{
  __shared__ float sw[512 * 8];
  __shared__ float sb[8];
  for (int i = threadIdx.x; i < 512 * 8; i += 256) sw[i] = w3[i];
  if (threadIdx.x < 8) sb[threadIdx.x] = b3[threadIdx.x];
  __syncthreads();
  const int e = blockIdx.x * 256 + threadIdx.x;
  if (e >= M) return;
  const u16b* hp = h2 + (size_t)e * 512;
  float acc[8] = {0,0,0,0,0,0,0,0};
  for (int k = 0; k < 512; k += 8) {
    uint4 u = *(const uint4*)(hp + k);
    const u16b* hs = (const u16b*)&u;
    #pragma unroll
    for (int t = 0; t < 8; ++t) {
      const float x = bf2f(hs[t]);
      const float* wr = &sw[(k + t) * 8];
      #pragma unroll
      for (int h = 0; h < 8; ++h) acc[h] += x * wr[h];
    }
  }
  const int s = srcI[e];
  #pragma unroll
  for (int h = 0; h < 8; ++h) {
    const float v = acc[h] + sb[h];
    scores[(size_t)e * 8 + h] = v;
    atomicMax(&segK[(size_t)s * 8 + h], fkey(v));
  }
}

__global__ __launch_bounds__(256) void k_att_exp(const float* scores, const unsigned* segK,
    const int* srcI, float* att, float* segS, int M)
{
  const int idx = blockIdx.x * 256 + threadIdx.x;
  if (idx >= M * 8) return;
  const int e = idx >> 3, h = idx & 7;
  const int s = srcI[e];
  const float m = fdec(segK[(size_t)s * 8 + h]);
  const float v = __expf(scores[idx] - m);
  att[idx] = v;
  atomicAdd(&segS[(size_t)s * 8 + h], v);
}

__global__ __launch_bounds__(256) void k_att_norm(float* att, const float* segS,
    const int* srcI, int M)
{
  const int idx = blockIdx.x * 256 + threadIdx.x;
  if (idx >= M * 8) return;
  const int e = idx >> 3, h = idx & 7;
  att[idx] = att[idx] / segS[(size_t)srcI[e] * 8 + h];
}

// out[src] += att[e,:] * v[dst,c]  (msg = att[:,None,:] * v[dst][:,:,None], reshape c*8+h)
__global__ __launch_bounds__(256) void k_scatter(const float* __restrict__ att,
    const float* __restrict__ vmat, const int* __restrict__ srcI,
    const int* __restrict__ dstI, float* __restrict__ accm, int M)
{
  const long long idx = (long long)blockIdx.x * 256 + threadIdx.x;
  if (idx >= (long long)M * 64) return;
  const int e = (int)(idx >> 6), c = (int)(idx & 63);
  const int s = srcI[e], d = dstI[e];
  const float vv = vmat[(size_t)d * 64 + c];
  const float4* ap = (const float4*)(att + (size_t)e * 8);
  const float4 a0 = ap[0], a1 = ap[1];
  float* dst = accm + (size_t)s * 512 + (size_t)c * 8;
  atomicAdd(dst + 0, a0.x * vv); atomicAdd(dst + 1, a0.y * vv);
  atomicAdd(dst + 2, a0.z * vv); atomicAdd(dst + 3, a0.w * vv);
  atomicAdd(dst + 4, a1.x * vv); atomicAdd(dst + 5, a1.y * vv);
  atomicAdd(dst + 6, a1.z * vv); atomicAdd(dst + 7, a1.w * vv);
}

__global__ __launch_bounds__(256) void k_update_atom(const float* a, const float* s,
    float* prev, float* atom, int res, size_t n)
{
  size_t i = (size_t)blockIdx.x * 256 + threadIdx.x;
  if (i >= n) return;
  float t = a[i] + s[i];
  if (res) { t += prev[i]; t = t > 0.f ? t : 0.f; prev[i] = t; }
  atom[i] = t;
}

__global__ __launch_bounds__(256) void k_mask_out(float* atom, const float* mask,
    float* outNS, size_t n)
{
  size_t i = (size_t)blockIdx.x * 256 + threadIdx.x;
  if (i >= n) return;
  const float v = atom[i] * mask[i / 512];
  atom[i] = v; outNS[i] = v;
}

// ---------------- host orchestration ----------------
static inline size_t cdiv(size_t a, size_t b){ return (a + b - 1) / b; }

static void launch_gemm(hipStream_t st, int mode,
    const u16b* A0, const u16b* A1, const int* i0, const int* i1, const int* i2,
    const u16b* B, const float* bias, float* oF, u16b* oB,
    int M, int N, int K, int segW, int act)
{
  dim3 g((unsigned)cdiv(N, 64), (unsigned)cdiv(M, 128)), b(256);
  if (mode == 0)      gemm_wmma<0><<<g, b, 0, st>>>(A0,A1,i0,i1,i2,B,bias,oF,oB,M,N,K,segW,act);
  else if (mode == 1) gemm_wmma<1><<<g, b, 0, st>>>(A0,A1,i0,i1,i2,B,bias,oF,oB,M,N,K,segW,act);
  else                gemm_wmma<2><<<g, b, 0, st>>>(A0,A1,i0,i1,i2,B,bias,oF,oB,M,N,K,segW,act);
}
static void ew_fill_f32(hipStream_t st, float* p, float v, size_t n){
  k_fill_f32<<<(unsigned)cdiv(n,256),256,0,st>>>(p,v,n); }
static void ew_fill_u32(hipStream_t st, unsigned* p, unsigned v, size_t n){
  k_fill_u32<<<(unsigned)cdiv(n,256),256,0,st>>>(p,v,n); }
static void ew_f2bf(hipStream_t st, const float* s, u16b* d, size_t n){
  k_f2bf<<<(unsigned)cdiv(n,256),256,0,st>>>(s,d,n); }

extern "C" void kernel_launch(void* const* d_in, const int* in_sizes, int n_in,
                              void* d_out, int out_size, void* d_ws, size_t ws_size,
                              hipStream_t stream)
{
  (void)in_sizes; (void)n_in; (void)out_size; (void)ws_size;
  // ---- inputs (setup_inputs() dict order, params flattened in insertion order) ----
  const float* in_atom  = (const float*)d_in[0];
  const float* in_bond  = (const float*)d_in[1];
  const int*   edge_idx = (const int*)  d_in[2];
  const int*   edge_sub = (const int*)  d_in[3];
  /* graph_id d_in[4] unused */
  const float* atom_mask= (const float*)d_in[5];
  const float* v_w   = (const float*)d_in[6];   const float* v_b   = (const float*)d_in[7];
  const float* a1w   = (const float*)d_in[8];   const float* a1b   = (const float*)d_in[9];
  const float* a2w   = (const float*)d_in[10];  const float* a2b   = (const float*)d_in[11];
  const float* a3w   = (const float*)d_in[12];  const float* a3b   = (const float*)d_in[13];
  const float* cw    = (const float*)d_in[14];  const float* cb    = (const float*)d_in[15];
  const float* bw    = (const float*)d_in[16];  const float* bb    = (const float*)d_in[17];
  const float* fAw0  = (const float*)d_in[18];  const float* fAb0  = (const float*)d_in[19];
  const float* fAw1  = (const float*)d_in[20];  const float* fAb1  = (const float*)d_in[21];
  const float* fAw2  = (const float*)d_in[22];  const float* fAb2  = (const float*)d_in[23];
  const float* fBw0  = (const float*)d_in[24];  const float* fBb0  = (const float*)d_in[25];
  const float* fBw1  = (const float*)d_in[26];  const float* fBb1  = (const float*)d_in[27];
  const float* fBw2  = (const float*)d_in[28];  const float* fBb2  = (const float*)d_in[29];
  const float* apW   = (const float*)d_in[30];  const float* apBb  = (const float*)d_in[31];
  const float* bpW   = (const float*)d_in[32];  const float* bpBb  = (const float*)d_in[33];
  float* out = (float*)d_out;

  // ---- workspace bump allocator ----
  char* base = (char*)d_ws; size_t off = 0;
  auto alloc = [&](size_t bytes)->void* {
    void* r = base + off; off = (off + bytes + 255) & ~(size_t)255; return r; };

  float* atomF = (float*)alloc((size_t)cN*cD*4);
  float* prevF = (float*)alloc((size_t)cN*cD*4);
  float* aF    = (float*)alloc((size_t)cN*cD*4);
  float* sF    = (float*)alloc((size_t)cN*cD*4);
  float* accF  = (float*)alloc((size_t)cN*cD*4);
  float* bond0 = (float*)alloc((size_t)cE*cD*4);
  float* bond1 = (float*)alloc((size_t)cE*cD*4);
  u16b* atomB  = (u16b*)alloc((size_t)cN*cD*2);
  u16b* outPreB= (u16b*)alloc((size_t)cN*cD*2);
  u16b* bondB  = (u16b*)alloc((size_t)cE*cD*2);
  u16b* nbB    = (u16b*)alloc((size_t)cE*cD*2);
  u16b* h1B    = (u16b*)alloc((size_t)cE*cD*2);
  u16b* h2B    = (u16b*)alloc((size_t)cE*cD*2);
  float* vF    = (float*)alloc((size_t)cN*64*4);
  float* scores= (float*)alloc((size_t)cE*8*4);
  float* att   = (float*)alloc((size_t)cE*8*4);
  unsigned* segK = (unsigned*)alloc((size_t)cN*8*4);
  float* segS  = (float*)alloc((size_t)cN*8*4);
  int* esrc = (int*)alloc((size_t)cE*4);
  int* edst = (int*)alloc((size_t)cE*4);
  int* ssrc = (int*)alloc((size_t)cES*4);
  int* sdst = (int*)alloc((size_t)cES*4);
  // bf16 weights
  u16b* vwB  = (u16b*)alloc((size_t)cL*cD*64*2);
  u16b* w1B  = (u16b*)alloc((size_t)cL*3*cD*cD*2);
  u16b* w2B  = (u16b*)alloc((size_t)cL*cD*cD*2);
  u16b* cwB  = (u16b*)alloc((size_t)cL*cD*cD*2);
  u16b* bwB  = (u16b*)alloc((size_t)cL*2*cD*cD*2);
  u16b* fAw0B= (u16b*)alloc((size_t)cD*FCH*2);
  u16b* fAw1B= (u16b*)alloc((size_t)FCH*FCH*2);
  u16b* fAw2B= (u16b*)alloc((size_t)FCH*NAo*2);
  u16b* fBw0B= (u16b*)alloc((size_t)2*cBA*FCH*2);
  u16b* fBw1B= (u16b*)alloc((size_t)FCH*FCH*2);
  u16b* fBw2B= (u16b*)alloc((size_t)FCH*NBo*2);
  u16b* apWB = (u16b*)alloc((size_t)cD*cBA*2);
  u16b* bpWB = (u16b*)alloc((size_t)cD*cBA*2);
  u16b* apB  = (u16b*)alloc((size_t)cN*cBA*2);
  u16b* bpB  = (u16b*)alloc((size_t)cE*cBA*2);
  u16b* t1B  = (u16b*)alloc((size_t)cN*FCH*2);
  u16b* t2B  = (u16b*)alloc((size_t)cN*FCH*2);
  u16b* e1B  = (u16b*)alloc((size_t)cE*FCH*2);
  u16b* e2B  = (u16b*)alloc((size_t)cE*FCH*2);

  // ---- setup: copy state, build edge arrays, convert weights ----
  hipMemcpyAsync(atomF, in_atom, (size_t)cN*cD*4, hipMemcpyDeviceToDevice, stream);
  hipMemcpyAsync(prevF, in_atom, (size_t)cN*cD*4, hipMemcpyDeviceToDevice, stream);
  hipMemcpyAsync(bond0, in_bond, (size_t)cE*cD*4, hipMemcpyDeviceToDevice, stream);
  k_build_edges<<<(unsigned)cdiv(cE,256),256,0,stream>>>(edge_idx, esrc, edst, cE);
  k_build_sub  <<<(unsigned)cdiv(cES,256),256,0,stream>>>(edge_idx, edge_sub, ssrc, sdst, cES);
  ew_f2bf(stream, v_w,  vwB,  (size_t)cL*cD*64);
  ew_f2bf(stream, a1w,  w1B,  (size_t)cL*3*cD*cD);
  ew_f2bf(stream, a2w,  w2B,  (size_t)cL*cD*cD);
  ew_f2bf(stream, cw,   cwB,  (size_t)cL*cD*cD);
  ew_f2bf(stream, bw,   bwB,  (size_t)cL*2*cD*cD);
  ew_f2bf(stream, fAw0, fAw0B,(size_t)cD*FCH);
  ew_f2bf(stream, fAw1, fAw1B,(size_t)FCH*FCH);
  ew_f2bf(stream, fAw2, fAw2B,(size_t)FCH*NAo);
  ew_f2bf(stream, fBw0, fBw0B,(size_t)2*cBA*FCH);
  ew_f2bf(stream, fBw1, fBw1B,(size_t)FCH*FCH);
  ew_f2bf(stream, fBw2, fBw2B,(size_t)FCH*NBo);
  ew_f2bf(stream, apW,  apWB, (size_t)cD*cBA);
  ew_f2bf(stream, bpW,  bpWB, (size_t)cD*cBA);

  // shared conv body: attention softmax + message pass + conv GEMM
  auto run_conv = [&](int Me, const int* srcI, const int* dstI, const int* bondRowIdx,
                      const u16b* atomInB, const u16b* bondSrcB,
                      const u16b* lw1B, const float* lb1,
                      const u16b* lw2B, const float* lb2,
                      const float* lw3, const float* lb3,
                      const u16b* lvwB, const float* lvb,
                      const u16b* lcwB, const float* lcb,
                      int convAct, float* outAtomF, u16b* outAtomPreB)
  {
    // h1 = relu(concat(atom[dst], atom[src], bond) @ w1 + b1)
    launch_gemm(stream, 1, atomInB, bondSrcB, dstI, srcI, bondRowIdx,
                lw1B, lb1, nullptr, h1B, Me, cD, 3*cD, cD, 1);
    // h2 = relu(h1 @ w2 + b2)
    launch_gemm(stream, 0, h1B, nullptr, nullptr, nullptr, nullptr,
                lw2B, lb2, nullptr, h2B, Me, cD, cD, 0, 1);
    // scatter softmax by src
    ew_fill_u32(stream, segK, 0u, (size_t)cN*8);
    ew_fill_f32(stream, segS, 0.f, (size_t)cN*8);
    k_scores  <<<(unsigned)cdiv(Me,256),256,0,stream>>>(h2B, lw3, lb3, srcI, scores, segK, Me);
    k_att_exp <<<(unsigned)cdiv((size_t)Me*8,256),256,0,stream>>>(scores, segK, srcI, att, segS, Me);
    k_att_norm<<<(unsigned)cdiv((size_t)Me*8,256),256,0,stream>>>(att, segS, srcI, Me);
    // v = atom @ v_w + v_b
    launch_gemm(stream, 0, atomInB, nullptr, nullptr, nullptr, nullptr,
                lvwB, lvb, vF, nullptr, cN, 64, cD, 0, 0);
    // segment-sum of att * v[dst]
    ew_fill_f32(stream, accF, 0.f, (size_t)cN*cD);
    k_scatter<<<(unsigned)cdiv((size_t)Me*64,256),256,0,stream>>>(att, vF, srcI, dstI, accF, Me);
    // out = acc @ conv_w + conv_b  (accB reuses h1B)
    ew_f2bf(stream, accF, h1B, (size_t)cN*cD);
    launch_gemm(stream, 0, h1B, nullptr, nullptr, nullptr, nullptr,
                lcwB, lcb, outAtomF, outAtomPreB, cN, cD, cD, 0, convAct);
  };

  float* bondCur = bond0;
  float* bondNxt = bond1;
  for (int i = 0; i < cL; ++i) {
    const int res = (i % 2 == 1);
    const int act = !res;
    const size_t oD = (size_t)i * cD * cD;
    const u16b* lw1 = w1B + (size_t)i*3*cD*cD; const float* lb1 = a1b + (size_t)i*cD;
    const u16b* lw2 = w2B + oD;                const float* lb2 = a2b + (size_t)i*cD;
    const float* lw3 = a3w + (size_t)i*cD*8;   const float* lb3 = a3b + (size_t)i*8;
    const u16b* lvw = vwB + (size_t)i*cD*64;   const float* lvb = v_b + (size_t)i*64;
    const u16b* lcw = cwB + oD;                const float* lcb = cb  + (size_t)i*cD;
    const u16b* lbw = bwB + (size_t)i*2*cD*cD; const float* lbb = bb  + (size_t)i*cD;

    // ---- conv 1 (full edge set) ----
    ew_f2bf(stream, atomF, atomB, (size_t)cN*cD);
    ew_f2bf(stream, bondCur, bondB, (size_t)cE*cD);
    run_conv(cE, esrc, edst, nullptr, atomB, bondB,
             lw1, lb1, lw2, lb2, lw3, lb3, lvw, lvb, lcw, lcb,
             /*convAct=*/0, aF, outPreB);           // pre-activation out kept for nb
    // nb = relu(concat(out[dst]+out[src], bond) @ bond_w + bond_b)
    launch_gemm(stream, 2, outPreB, bondB, edst, esrc, nullptr,
                lbw, lbb, bondNxt, nbB, cE, cD, 2*cD, cD, 1);
    if (act) k_relu<<<(unsigned)cdiv((size_t)cN*cD,256),256,0,stream>>>(aF, (size_t)cN*cD);

    // ---- conv 2 (sub edge set, bond = nb[edge_sub]) ----
    ew_f2bf(stream, aF, atomB, (size_t)cN*cD);
    run_conv(cES, ssrc, sdst, edge_sub, atomB, nbB,
             lw1, lb1, lw2, lb2, lw3, lb3, lvw, lvb, lcw, lcb,
             /*convAct=*/act ? 1 : 0, sF, nullptr);

    // atom = a + s ; if res: atom = relu(atom + prev), prev = atom ; bond = nb
    k_update_atom<<<(unsigned)cdiv((size_t)cN*cD,256),256,0,stream>>>(
        aF, sF, prevF, atomF, res, (size_t)cN*cD);
    float* t = bondCur; bondCur = bondNxt; bondNxt = t;
  }

  // ---- heads ----
  // node_state = mask * atom  ->  out[0 : N*D)
  k_mask_out<<<(unsigned)cdiv((size_t)cN*cD,256),256,0,stream>>>(
      atomF, atom_mask, out, (size_t)cN*cD);
  ew_f2bf(stream, atomF, atomB, (size_t)cN*cD);
  // fc_atom: 512->256 relu, 256->256 relu, 256->100  -> out[N*D : N*D+N*100)
  launch_gemm(stream, 0, atomB, nullptr, nullptr, nullptr, nullptr,
              fAw0B, fAb0, nullptr, t1B, cN, FCH, cD, 0, 1);
  launch_gemm(stream, 0, t1B, nullptr, nullptr, nullptr, nullptr,
              fAw1B, fAb1, nullptr, t2B, cN, FCH, FCH, 0, 1);
  launch_gemm(stream, 0, t2B, nullptr, nullptr, nullptr, nullptr,
              fAw2B, fAb2, out + (size_t)cN*cD, nullptr, cN, NAo, FCH, 0, 0);
  // ap = tanh(atom @ atom_proj), bp = tanh(bond @ bond_proj)
  launch_gemm(stream, 0, atomB, nullptr, nullptr, nullptr, nullptr,
              apWB, apBb, nullptr, apB, cN, cBA, cD, 0, 2);
  ew_f2bf(stream, bondCur, bondB, (size_t)cE*cD);
  launch_gemm(stream, 0, bondB, nullptr, nullptr, nullptr, nullptr,
              bpWB, bpBb, nullptr, bpB, cE, cBA, cD, 0, 2);
  // bf = concat(ap[e0]+ap[e1], bp) -> 256 relu -> 256 relu -> 50
  launch_gemm(stream, 2, apB, bpB, esrc, edst, nullptr,
              fBw0B, fBb0, nullptr, e1B, cE, FCH, 2*cBA, cBA, 1);
  launch_gemm(stream, 0, e1B, nullptr, nullptr, nullptr, nullptr,
              fBw1B, fBb1, nullptr, e2B, cE, FCH, FCH, 0, 1);
  launch_gemm(stream, 0, e2B, nullptr, nullptr, nullptr, nullptr,
              fBw2B, fBb2, out + (size_t)cN*cD + (size_t)cN*NAo, nullptr, cE, NBo, FCH, 0, 0);
}
